// SelfAttentionHead_56349970923690
// MI455X (gfx1250) — matrix operand order repaired and checked by reading, not compiled, not run on hardware
//
#include <hip/hip_runtime.h>
#include <hip/hip_bf16.h>

typedef _Float16 h16;
typedef __attribute__((ext_vector_type(16))) _Float16 v16h;
typedef __attribute__((ext_vector_type(8)))  _Float16 v8h;
typedef __attribute__((ext_vector_type(8)))  float    v8f;
typedef __attribute__((ext_vector_type(4)))  float    v4f;

constexpr int kB = 4, kT = 2048, kC = 1024, kH = 64;

static __device__ __forceinline__ v8f wmma_f16(v16h a, v16h b, v8f c) {
  return __builtin_amdgcn_wmma_f32_16x16x32_f16(false, a, false, b, (short)0, c,
                                                false, false);
}

static __device__ __forceinline__ v16h cat8(v8h lo, v8h hi) {
  v16h r;
#pragma unroll
  for (int i = 0; i < 8; ++i) { r[i] = lo[i]; r[i + 8] = hi[i]; }
  return r;
}

// A-matrix 16x32 f16 slice at (row0, k0) from row-major [rows, ld] f16.
// Lane layout: row = lane&15; elems 0..15 -> k0 + 16*hf + 0..15 (contiguous).
static __device__ __forceinline__ v16h loadA(const h16* base, int ld, int row0,
                                             int k0, int l16, int hf) {
  const h16* p = base + (size_t)(row0 + l16) * ld + k0 + 16 * hf;
  v8h lo = *(const v8h*)p;
  v8h hi = *(const v8h*)(p + 8);
  return cat8(lo, hi);
}

// Same A slice but from f32 source, converting to f16 on the fly.
static __device__ __forceinline__ v16h loadA_f32(const float* base, int ld,
                                                 int row0, int k0, int l16,
                                                 int hf) {
  const float* p = base + (size_t)(row0 + l16) * ld + k0 + 16 * hf;
  v4f a0 = *(const v4f*)p;
  v4f a1 = *(const v4f*)(p + 4);
  v4f b0 = *(const v4f*)(p + 8);
  v4f b1 = *(const v4f*)(p + 12);
  v16h r;
#pragma unroll
  for (int i = 0; i < 4; ++i) {
    r[i]      = (h16)a0[i];
    r[i + 4]  = (h16)a1[i];
    r[i + 8]  = (h16)b0[i];
    r[i + 12] = (h16)b1[i];
  }
  return r;
}

// B-matrix 32x16 f16: element (k,n) at base[(n0+n)*ld + k0 + k].
// Lane layout: n = lane&15; elems 0..15 -> k = 16*hf + 0..15 (contiguous).
static __device__ __forceinline__ v16h loadB(const h16* base, int ld, int n0,
                                             int k0, int l16, int hf) {
  const h16* p = base + (size_t)(n0 + l16) * ld + k0 + 16 * hf;
  v8h lo = *(const v8h*)p;
  v8h hi = *(const v8h*)(p + 8);
  return cat8(lo, hi);
}

static __device__ __forceinline__ float rmax16(float v) {
#pragma unroll
  for (int off = 1; off < 16; off <<= 1) v = fmaxf(v, __shfl_xor(v, off, 32));
  return v;
}
static __device__ __forceinline__ float rsum16(float v) {
#pragma unroll
  for (int off = 1; off < 16; off <<= 1) v += __shfl_xor(v, off, 32);
  return v;
}

// ---------------------------------------------------------------------------
// Kernel 1: transpose+convert weights to f16 for contiguous-K B-matrix loads.
//   Wq/Wk/Wv [C,H] f32  -> WT [H,C] f16
//   Wp       [H,C] f32  -> WpT [C,H] f16
// ---------------------------------------------------------------------------
__global__ __launch_bounds__(256) void wprep_kernel(
    const float* __restrict__ Wq, const float* __restrict__ Wk,
    const float* __restrict__ Wv, const float* __restrict__ Wp,
    h16* __restrict__ WqT, h16* __restrict__ WkT, h16* __restrict__ WvT,
    h16* __restrict__ WpT) {
  const int id = blockIdx.x * 256 + threadIdx.x;  // 0 .. C*H-1
  const int which = blockIdx.y;
  if (which < 3) {
    const float* src = (which == 0) ? Wq : (which == 1) ? Wk : Wv;
    h16* dst = (which == 0) ? WqT : (which == 1) ? WkT : WvT;
    const int c = id / kH, h = id % kH;
    dst[(size_t)h * kC + c] = (h16)src[(size_t)c * kH + h];
  } else {
    const int h = id / kC, c = id % kC;
    WpT[(size_t)c * kH + h] = (h16)Wp[(size_t)h * kC + c];
  }
}

// ---------------------------------------------------------------------------
// Kernel 2: QKV projection. One wave per (16-row tile, projection) so the
// grid carries 1536 light waves (32 accumulator VGPRs each) and load waits
// are hidden by wave switching; x re-reads stay L2-resident (192 MB L2).
// Writes Q,K as [B,T,H] f16 and V transposed as [B,H,T] f16.
// ---------------------------------------------------------------------------
__global__ __launch_bounds__(32) void qkv_kernel(
    const float* __restrict__ x, const h16* __restrict__ WqT,
    const h16* __restrict__ WkT, const h16* __restrict__ WvT,
    const float* __restrict__ bq, const float* __restrict__ bk,
    const float* __restrict__ bv, h16* __restrict__ Qh, h16* __restrict__ Kh,
    h16* __restrict__ VTh) {
  const int lane = threadIdx.x & 31, hf = lane >> 4, l16 = lane & 15;
  const int g = blockIdx.x;   // row tile
  const int p = blockIdx.y;   // 0=q, 1=k, 2=v
  const int b = (g * 16) / kT;
  const int t0 = (g * 16) % kT;
  const float* xb = x + (size_t)b * kT * kC;
  const h16* WT = (p == 0) ? WqT : (p == 1) ? WkT : WvT;
  const float* bias = (p == 0) ? bq : (p == 1) ? bk : bv;

  v8f acc[4];
#pragma unroll
  for (int n = 0; n < 4; ++n) acc[n] = {0, 0, 0, 0, 0, 0, 0, 0};

  v16h a = loadA_f32(xb, kC, t0, 0, l16, hf);
  for (int kc = 0; kc < kC; kc += 32) {
    v16h bm[4];
#pragma unroll
    for (int n = 0; n < 4; ++n) bm[n] = loadB(WT, kC, n * 16, kc, l16, hf);
    // Pipeline next A tile (clamped re-load on the final step; stays in-row).
    const int kn = (kc + 32 < kC) ? (kc + 32) : kc;
    const v16h a_next = loadA_f32(xb, kC, t0, kn, l16, hf);
#pragma unroll
    for (int n = 0; n < 4; ++n) acc[n] = wmma_f16(a, bm[n], acc[n]);
    a = a_next;
  }

  if (p == 2) {
    h16* Vb = VTh + (size_t)b * kH * kT;  // store transposed [H,T]
#pragma unroll
    for (int n = 0; n < 4; ++n) {
      const int col = n * 16 + l16;
      const float bv_ = bias[col];
#pragma unroll
      for (int r = 0; r < 8; ++r) {
        const int m = r + 8 * hf;
        Vb[(size_t)col * kT + (t0 + m)] = (h16)(acc[n][r] + bv_);
      }
    }
  } else {
    h16* Db = ((p == 0) ? Qh : Kh) + (size_t)b * kT * kH;
#pragma unroll
    for (int n = 0; n < 4; ++n) {
      const int col = n * 16 + l16;
      const float bv_ = bias[col];
#pragma unroll
      for (int r = 0; r < 8; ++r) {
        const int m = r + 8 * hf;
        Db[(size_t)(t0 + m) * kH + col] = (h16)(acc[n][r] + bv_);
      }
    }
  }
}

// ---------------------------------------------------------------------------
// Kernel 3: flash-attention core, 4 cooperating waves per (batch, q-tile).
// Wave w handles every 4th 32-column kv step with private online-softmax
// state and a private LDS P-slice; partial (m, l, O) are combined across
// waves through LDS at the end (flash-decoding style). Critical path /4.
// ---------------------------------------------------------------------------
__global__ __launch_bounds__(128) void attn_kernel(const h16* __restrict__ Qh,
                                                   const h16* __restrict__ Kh,
                                                   const h16* __restrict__ VTh,
                                                   h16* __restrict__ Oh) {
  __shared__ __align__(16) h16 pT[4][16 * 32];
  __shared__ float cm[4][16];
  __shared__ float cl[4][16];
  __shared__ float cO[4][16][kH];
  const int w = threadIdx.x >> 5;  // wave id 0..3
  const int lane = threadIdx.x & 31, hf = lane >> 4, l16 = lane & 15;
  const int nqt = kT / 16;
  const int b = blockIdx.x / nqt;
  const int i = blockIdx.x % nqt;
  const int tq = i * 16;
  const h16* Qb = Qh + (size_t)b * kT * kH;
  const h16* Kb = Kh + (size_t)b * kT * kH;
  const h16* Vb = VTh + (size_t)b * kH * kT;
  h16* myP = &pT[w][0];

  const v16h a0 = loadA(Qb, kH, tq, 0, l16, hf);
  const v16h a1 = loadA(Qb, kH, tq, 32, l16, hf);

  const float NEG = -__builtin_inff();
  const float scale = 0.03125f;  // 1/sqrt(C), matching the reference
  float mrow[8], lrow[8], alc[8];
  v8f o[4];
#pragma unroll
  for (int r = 0; r < 8; ++r) { mrow[r] = NEG; lrow[r] = 0.f; }
#pragma unroll
  for (int n = 0; n < 4; ++n) o[n] = {0, 0, 0, 0, 0, 0, 0, 0};

  for (int j0 = 2 * w; j0 <= i; j0 += 8) {  // 4 waves stride the kv range
    const int j1 = j0 + 1;
    const bool have1 = (j1 <= i);

    // S = Q K^T for two 16-column kv tiles (K split 0..31 / 32..63).
    v8f s0 = {0, 0, 0, 0, 0, 0, 0, 0};
    s0 = wmma_f16(a0, loadB(Kb, kH, j0 * 16, 0, l16, hf), s0);
    s0 = wmma_f16(a1, loadB(Kb, kH, j0 * 16, 32, l16, hf), s0);
    v8f s1 = {0, 0, 0, 0, 0, 0, 0, 0};
    if (have1) {
      s1 = wmma_f16(a0, loadB(Kb, kH, j1 * 16, 0, l16, hf), s1);
      s1 = wmma_f16(a1, loadB(Kb, kH, j1 * 16, 32, l16, hf), s1);
    }

    // Issue V B-tile loads now; independent of the softmax math below.
    v16h bvv[4];
#pragma unroll
    for (int n = 0; n < 4; ++n) {
      v16h z = {0, 0, 0, 0, 0, 0, 0, 0, 0, 0, 0, 0, 0, 0, 0, 0};
      bvv[n] = z;
      if (have1 || hf == 0)  // hf==1 lanes hold kv cols 16..31 of this step
        bvv[n] = loadB(Vb, kT, n * 16, j0 * 16, l16, hf);
    }

#pragma unroll
    for (int r = 0; r < 8; ++r) {
      const int m = r + 8 * hf;
      float v0 = s0[r] * scale;
      if (j0 == i && l16 > m) v0 = NEG;        // causal mask, diagonal tile
      float v1 = NEG;
      if (have1) {
        v1 = s1[r] * scale;
        if (j1 == i && l16 > m) v1 = NEG;
      }
      const float mx = rmax16(fmaxf(v0, v1));  // row max across 16 lanes
      const float nm = fmaxf(mrow[r], mx);
      const float al = __expf(mrow[r] - nm);
      const float p0 = __expf(v0 - nm);
      const float p1 = __expf(v1 - nm);
      lrow[r] = lrow[r] * al + rsum16(p0 + p1);
      mrow[r] = nm;
      alc[r] = al;
      myP[m * 32 + l16] = (h16)p0;
      myP[m * 32 + 16 + l16] = (h16)p1;
    }
#pragma unroll
    for (int n = 0; n < 4; ++n) {
#pragma unroll
      for (int r = 0; r < 8; ++r) o[n][r] *= alc[r];
    }
    const v16h ap = loadA(myP, 32, 0, 0, l16, hf);  // P in A layout (K=32)
#pragma unroll
    for (int n = 0; n < 4; ++n) o[n] = wmma_f16(ap, bvv[n], o[n]);
  }

  // Publish this wave's partial state.
#pragma unroll
  for (int r = 0; r < 8; ++r) {
    const int m = r + 8 * hf;
    if (l16 == 0) { cm[w][m] = mrow[r]; cl[w][m] = lrow[r]; }
#pragma unroll
    for (int n = 0; n < 4; ++n) cO[w][m][n * 16 + l16] = o[n][r];
  }
  __syncthreads();

  // Cross-wave combine; wave w produces output columns [16w, 16w+16).
  h16* Ob = Oh + (size_t)b * kT * kH;
#pragma unroll
  for (int r = 0; r < 8; ++r) {
    const int m = r + 8 * hf;
    float mf = cm[0][m];
#pragma unroll
    for (int w2 = 1; w2 < 4; ++w2) mf = fmaxf(mf, cm[w2][m]);
    float lsum = 0.f, osum = 0.f;
#pragma unroll
    for (int w2 = 0; w2 < 4; ++w2) {
      const float f = __expf(cm[w2][m] - mf);
      lsum += f * cl[w2][m];
      osum += f * cO[w2][m][w * 16 + l16];
    }
    Ob[(size_t)(tq + m) * kH + w * 16 + l16] = (h16)(osum / lsum);
  }
}

// ---------------------------------------------------------------------------
// Kernel 4: output projection  out[B*T, C] = O[B*T, H] @ Wp[H, C] + bp (f32).
// One wave per (16-row tile, 256-column chunk) -> 2048 waves.
// ---------------------------------------------------------------------------
__global__ __launch_bounds__(32) void proj_kernel(const h16* __restrict__ Oh,
                                                  const h16* __restrict__ WpT,
                                                  const float* __restrict__ bp,
                                                  float* __restrict__ out) {
  const int lane = threadIdx.x & 31, hf = lane >> 4, l16 = lane & 15;
  const int row0 = (blockIdx.x >> 2) * 16;
  const int nbase = (blockIdx.x & 3) * 16;  // first of 16 column tiles
  const v16h a0 = loadA(Oh, kH, row0, 0, l16, hf);
  const v16h a1 = loadA(Oh, kH, row0, 32, l16, hf);
  for (int n = nbase; n < nbase + 16; n += 2) {
    // Two output tiles per step: batch 8 B-loads, then 4 WMMAs.
    v16h b00 = loadB(WpT, kH, n * 16, 0, l16, hf);
    v16h b01 = loadB(WpT, kH, n * 16, 32, l16, hf);
    v16h b10 = loadB(WpT, kH, (n + 1) * 16, 0, l16, hf);
    v16h b11 = loadB(WpT, kH, (n + 1) * 16, 32, l16, hf);
    v8f acc0 = {0, 0, 0, 0, 0, 0, 0, 0};
    v8f acc1 = {0, 0, 0, 0, 0, 0, 0, 0};
    acc0 = wmma_f16(a0, b00, acc0);
    acc0 = wmma_f16(a1, b01, acc0);
    acc1 = wmma_f16(a0, b10, acc1);
    acc1 = wmma_f16(a1, b11, acc1);
    const int col0 = n * 16 + l16;
    const int col1 = col0 + 16;
    const float bias0 = bp[col0];
    const float bias1 = bp[col1];
#pragma unroll
    for (int r = 0; r < 8; ++r) {
      const size_t rowoff = (size_t)(row0 + r + 8 * hf) * kC;
      out[rowoff + col0] = acc0[r] + bias0;
      out[rowoff + col1] = acc1[r] + bias1;
    }
  }
}

extern "C" void kernel_launch(void* const* d_in, const int* in_sizes, int n_in,
                              void* d_out, int out_size, void* d_ws,
                              size_t ws_size, hipStream_t stream) {
  (void)in_sizes; (void)n_in; (void)out_size; (void)ws_size;
  const float* x  = (const float*)d_in[0];
  const float* Wk = (const float*)d_in[1];
  const float* bk = (const float*)d_in[2];
  const float* Wq = (const float*)d_in[3];
  const float* bq = (const float*)d_in[4];
  const float* Wv = (const float*)d_in[5];
  const float* bv = (const float*)d_in[6];
  const float* Wp = (const float*)d_in[7];
  const float* bp = (const float*)d_in[8];
  float* out = (float*)d_out;

  // Workspace layout (f16): Q,K [B,T,H]; V^T [B,H,T]; O [B,T,H]; weights.
  const size_t nBTH = (size_t)kB * kT * kH;   // 524288 halves = 1 MB each
  const size_t nCH  = (size_t)kC * kH;        // 65536 halves = 128 KB each
  h16* Qh  = (h16*)d_ws;
  h16* Kh  = Qh + nBTH;
  h16* VTh = Kh + nBTH;
  h16* Oh  = VTh + nBTH;
  h16* WqT = Oh + nBTH;
  h16* WkT = WqT + nCH;
  h16* WvT = WkT + nCH;
  h16* WpT = WvT + nCH;                       // total ~4.5 MB

  const int ntiles = (kB * kT) / 16;          // 512 row tiles

  dim3 gw((kC * kH) / 256, 4);
  wprep_kernel<<<gw, 256, 0, stream>>>(Wq, Wk, Wv, Wp, WqT, WkT, WvT, WpT);
  dim3 gqkv(ntiles, 3);
  qkv_kernel<<<gqkv, 32, 0, stream>>>(x, WqT, WkT, WvT, bq, bk, bv, Qh, Kh,
                                      VTh);
  attn_kernel<<<ntiles, 128, 0, stream>>>(Qh, Kh, VTh, Oh);
  proj_kernel<<<ntiles * 4, 32, 0, stream>>>(Oh, WpT, bp, out);
}
